// E_GCL_20607253086819
// MI455X (gfx1250) — matrix operand order repaired and checked by reading, not compiled, not run on hardware
//
#include <hip/hip_runtime.h>
#include <math.h>

#define NF 256
#define HF 256

typedef __attribute__((ext_vector_type(16))) __bf16 v16bf;
typedef __attribute__((ext_vector_type(8)))  __bf16 v8bf;
typedef __attribute__((ext_vector_type(8)))  float  v8f;

#define LDS_FENCE() asm volatile("s_wait_dscnt 0x0" ::: "memory")

__device__ __forceinline__ __bf16 f2bf(float f) {
  union { float f; unsigned u; } in; in.f = f;
  unsigned r = (in.u + 0x7FFFu + ((in.u >> 16) & 1u)) >> 16;  // RNE
  union { unsigned short s; __bf16 b; } out; out.s = (unsigned short)r;
  return out.b;
}

__device__ __forceinline__ float silu_f(float x) {
  return x / (1.0f + __expf(-x));
}

// Load one 16x32 bf16 A-fragment slice for this lane:
// lane<16 holds row m, K = {0..7, 16..23}; lane>=16 holds row m, K = {8..15, 24..31}.
// p points at (row base + ktile*32 + (half?8:0)); two contiguous 16B chunks.
template <typename P>
__device__ __forceinline__ v16bf ld_afrag(P p) {
  v8bf lo = *(const v8bf*)(p);
  v8bf hi = *(const v8bf*)(p + 16);
  v16bf a;
#pragma unroll
  for (int i = 0; i < 8; ++i) { a[i] = lo[i]; a[8 + i] = hi[i]; }
  return a;
}

__device__ __forceinline__ v8f wmma_bf16(v16bf a, v16bf b, v8f c) {
  return __builtin_amdgcn_wmma_f32_16x16x32_bf16(false, a, false, b, (short)0, c, false, false);
}

// ---------------------------------------------------------------- prep kernels
__global__ void zero_f32(float* p, int n) {
  int i = blockIdx.x * 256 + threadIdx.x;
  if (i < n) p[i] = 0.0f;
}

__global__ void cvt_bf16(const float* __restrict__ s, __bf16* __restrict__ d, int n) {
  int i = blockIdx.x * 256 + threadIdx.x;
  if (i < n) d[i] = f2bf(s[i]);
}

// Pack W [K x 256] row-major fp32 into B-fragment order:
// P[(((nt*KT + kt)*32 + lane)*16 + t)] = W[kt*32 + (lane>=16?8:0) + (t&7) + (t>=8?16:0)][nt*16 + (lane&15)]
__global__ void pack_w(const float* __restrict__ W, __bf16* __restrict__ P, int K) {
  int idx = blockIdx.x * 256 + threadIdx.x;
  int total = K * 256;
  if (idx >= total) return;
  int t    = idx & 15;
  int lane = (idx >> 4) & 31;
  int blk  = idx >> 9;
  int KT   = K >> 5;
  int kt   = blk % KT;
  int nt   = blk / KT;
  int k    = kt * 32 + ((lane >> 4) << 3) + (t & 7) + ((t >> 3) << 4);
  int n    = nt * 16 + (lane & 15);
  P[idx] = f2bf(W[k * 256 + n]);
}

// ---------------------------------------------------------------- edge kernel
__global__ __launch_bounds__(128)
void egcl_edge_kernel(const float* __restrict__ coord, const int* __restrict__ ei,
                      const __bf16* __restrict__ hbf,
                      const v16bf* __restrict__ We1p, const v16bf* __restrict__ We2p,
                      const v16bf* __restrict__ Wc1p,
                      const float* __restrict__ We1, const float* __restrict__ be1,
                      const float* __restrict__ be2, const float* __restrict__ bc1,
                      const float* __restrict__ Wc2, const float* __restrict__ bc2,
                      float* __restrict__ agg_h, float* __restrict__ agg_c,
                      float* __restrict__ cnt, int E, int ntiles)
{
  __shared__ __align__(16) __bf16 act[4][16 * HF];   // 8KB per wave
  __shared__ float sdiff[4][16 * 3];
  __shared__ float srad[4][16];
  __shared__ float ssum[4][16];
  __shared__ int   srow[4][16];

  const int wave = threadIdx.x >> 5;
  const int lane = threadIdx.x & 31;
  const int tile = blockIdx.x * 4 + wave;
  if (tile >= ntiles) return;

  const int m      = lane & 15;
  const int half   = lane >> 4;
  const int jbase  = half * 8;    // D rows this lane holds: jbase..jbase+7
  const int klaneA = half * 8;    // A-frag K sub-offset
  const int e = tile * 16 + m;
  const int r = ei[e];
  const int c = ei[E + e];

  if (lane < 16) {
    float dx = coord[r * 3 + 0] - coord[c * 3 + 0];
    float dy = coord[r * 3 + 1] - coord[c * 3 + 1];
    float dz = coord[r * 3 + 2] - coord[c * 3 + 2];
    sdiff[wave][m * 3 + 0] = dx;
    sdiff[wave][m * 3 + 1] = dy;
    sdiff[wave][m * 3 + 2] = dz;
    srad[wave][m] = dx * dx + dy * dy + dz * dz;
    srow[wave][m] = r;
  }
  LDS_FENCE();

  // ---- layer 1: [h_r | h_c | radial] @ We1 + be1, silu -> act (bf16 in LDS)
  v16bf a1[16];
  {
    const __bf16* hr = hbf + (size_t)r * NF;
    const __bf16* hc = hbf + (size_t)c * NF;
#pragma unroll
    for (int kt = 0; kt < 8; ++kt) a1[kt]     = ld_afrag(hr + kt * 32 + klaneA);
#pragma unroll
    for (int kt = 0; kt < 8; ++kt) a1[8 + kt] = ld_afrag(hc + kt * 32 + klaneA);
  }

#pragma unroll 1
  for (int nt = 0; nt < 16; ++nt) {
    const int n = nt * 16 + m;
    float bias = be1[n];
    v8f acc;
#pragma unroll
    for (int j = 0; j < 8; ++j) acc[j] = bias;
#pragma unroll
    for (int kt = 0; kt < 16; ++kt)
      acc = wmma_bf16(a1[kt], We1p[(nt * 16 + kt) * 32 + lane], acc);
    const float w512 = We1[512 * NF + n];   // radial column (K = 512), rank-1
#pragma unroll
    for (int j = 0; j < 8; ++j) {
      float x = acc[j] + srad[wave][jbase + j] * w512;
      act[wave][(jbase + j) * HF + n] = f2bf(silu_f(x));
    }
  }
  LDS_FENCE();

  // ---- layer 2: act @ We2 + be2, silu -> edge_feat; scatter to agg_h
  v16bf a2[8];
#pragma unroll
  for (int kt = 0; kt < 8; ++kt)
    a2[kt] = ld_afrag(&act[wave][m * HF + kt * 32 + klaneA]);
  LDS_FENCE();

#pragma unroll 1
  for (int nt = 0; nt < 16; ++nt) {
    const int n = nt * 16 + m;
    float bias = be2[n];
    v8f acc;
#pragma unroll
    for (int j = 0; j < 8; ++j) acc[j] = bias;
#pragma unroll
    for (int kt = 0; kt < 8; ++kt)
      acc = wmma_bf16(a2[kt], We2p[(nt * 8 + kt) * 32 + lane], acc);
#pragma unroll
    for (int j = 0; j < 8; ++j) {
      float ef = silu_f(acc[j]);
      unsafeAtomicAdd(&agg_h[(size_t)srow[wave][jbase + j] * HF + n], ef);
      act[wave][(jbase + j) * HF + n] = f2bf(ef);   // overwrite in place
    }
  }
  LDS_FENCE();

  // ---- layer 3: s = silu(ef @ Wc1 + bc1) @ Wc2 + bc2; coord scatter
  v16bf a3[8];
#pragma unroll
  for (int kt = 0; kt < 8; ++kt)
    a3[kt] = ld_afrag(&act[wave][m * HF + kt * 32 + klaneA]);

  float rowsum[8];
#pragma unroll
  for (int j = 0; j < 8; ++j) rowsum[j] = 0.0f;

#pragma unroll 1
  for (int nt = 0; nt < 16; ++nt) {
    const int n = nt * 16 + m;
    float bias = bc1[n];
    v8f acc;
#pragma unroll
    for (int j = 0; j < 8; ++j) acc[j] = bias;
#pragma unroll
    for (int kt = 0; kt < 8; ++kt)
      acc = wmma_bf16(a3[kt], Wc1p[(nt * 8 + kt) * 32 + lane], acc);
    const float wv = Wc2[n];
#pragma unroll
    for (int j = 0; j < 8; ++j) {
      float v = silu_f(acc[j]) * wv;
      v += __shfl_xor(v, 1, 16);
      v += __shfl_xor(v, 2, 16);
      v += __shfl_xor(v, 4, 16);
      v += __shfl_xor(v, 8, 16);
      rowsum[j] += v;                 // lanes with m==0 hold full 16-col sum
    }
  }
  if (m == 0) {
#pragma unroll
    for (int j = 0; j < 8; ++j) ssum[wave][jbase + j] = rowsum[j];
  }
  LDS_FENCE();

  if (lane < 16) {
    float s = ssum[wave][m] + bc2[0];
    float dx = sdiff[wave][m * 3 + 0];
    float dy = sdiff[wave][m * 3 + 1];
    float dz = sdiff[wave][m * 3 + 2];
    unsafeAtomicAdd(&agg_c[(size_t)r * 3 + 0], dx * s);
    unsafeAtomicAdd(&agg_c[(size_t)r * 3 + 1], dy * s);
    unsafeAtomicAdd(&agg_c[(size_t)r * 3 + 2], dz * s);
    unsafeAtomicAdd(&cnt[r], 1.0f);
  }
}

// ---------------------------------------------------------------- node kernel
__global__ __launch_bounds__(128)
void egcl_node_kernel(const __bf16* __restrict__ hbf, const float* __restrict__ agg_h,
                      const v16bf* __restrict__ Wn1p, const v16bf* __restrict__ Wn2p,
                      const float* __restrict__ bn1, const float* __restrict__ bn2,
                      float* __restrict__ h_out, int ntiles)
{
  __shared__ __align__(16) __bf16 act[4][16 * HF];
  const int wave = threadIdx.x >> 5;
  const int lane = threadIdx.x & 31;
  const int tile = blockIdx.x * 4 + wave;
  if (tile >= ntiles) return;
  const int m      = lane & 15;
  const int half   = lane >> 4;
  const int jbase  = half * 8;
  const int klaneA = half * 8;
  const int node = tile * 16 + m;

  v16bf a1[16];
#pragma unroll
  for (int kt = 0; kt < 8; ++kt)
    a1[kt] = ld_afrag(hbf + (size_t)node * NF + kt * 32 + klaneA);
#pragma unroll
  for (int kt = 0; kt < 8; ++kt) {
    const float* p = agg_h + (size_t)node * HF + kt * 32 + klaneA;
    v16bf a;
#pragma unroll
    for (int i = 0; i < 8; ++i) { a[i] = f2bf(p[i]); a[8 + i] = f2bf(p[16 + i]); }
    a1[8 + kt] = a;
  }

#pragma unroll 1
  for (int nt = 0; nt < 16; ++nt) {
    const int n = nt * 16 + m;
    float bias = bn1[n];
    v8f acc;
#pragma unroll
    for (int j = 0; j < 8; ++j) acc[j] = bias;
#pragma unroll
    for (int kt = 0; kt < 16; ++kt)
      acc = wmma_bf16(a1[kt], Wn1p[(nt * 16 + kt) * 32 + lane], acc);
#pragma unroll
    for (int j = 0; j < 8; ++j)
      act[wave][(jbase + j) * HF + n] = f2bf(silu_f(acc[j]));
  }
  LDS_FENCE();

  v16bf a2[8];
#pragma unroll
  for (int kt = 0; kt < 8; ++kt)
    a2[kt] = ld_afrag(&act[wave][m * HF + kt * 32 + klaneA]);

#pragma unroll 1
  for (int nt = 0; nt < 16; ++nt) {
    const int n = nt * 16 + m;
    float bias = bn2[n];
    v8f acc;
#pragma unroll
    for (int j = 0; j < 8; ++j) acc[j] = bias;
#pragma unroll
    for (int kt = 0; kt < 8; ++kt)
      acc = wmma_bf16(a2[kt], Wn2p[(nt * 8 + kt) * 32 + lane], acc);
#pragma unroll
    for (int j = 0; j < 8; ++j)
      h_out[(size_t)(tile * 16 + jbase + j) * NF + n] = acc[j];
  }
}

// ---------------------------------------------------------------- tails (scalar)
__global__ void egcl_edge_tail(const float* h, const float* coord, const int* ei,
                               const float* We1, const float* be1,
                               const float* We2, const float* be2,
                               const float* Wc1, const float* bc1,
                               const float* Wc2, const float* bc2,
                               float* agg_h, float* agg_c, float* cnt, int e0, int E)
{
  int e = e0 + blockIdx.x;
  if (e >= E) return;
  __shared__ float ein[2 * NF + 1];
  __shared__ float buf[HF];
  __shared__ float dxyz[3];
  int t = threadIdx.x;
  int r = ei[e], c = ei[E + e];
  ein[t]      = h[(size_t)r * NF + t];
  ein[NF + t] = h[(size_t)c * NF + t];
  if (t == 0) {
    float dx = coord[r * 3 + 0] - coord[c * 3 + 0];
    float dy = coord[r * 3 + 1] - coord[c * 3 + 1];
    float dz = coord[r * 3 + 2] - coord[c * 3 + 2];
    dxyz[0] = dx; dxyz[1] = dy; dxyz[2] = dz;
    ein[2 * NF] = dx * dx + dy * dy + dz * dz;
  }
  __syncthreads();
  float s = be1[t];
  for (int k = 0; k < 2 * NF + 1; ++k) s += ein[k] * We1[k * HF + t];
  float a1 = silu_f(s);
  __syncthreads(); buf[t] = a1; __syncthreads();
  float s2 = be2[t];
  for (int k = 0; k < HF; ++k) s2 += buf[k] * We2[k * HF + t];
  float ef = silu_f(s2);
  unsafeAtomicAdd(&agg_h[(size_t)r * HF + t], ef);
  __syncthreads(); buf[t] = ef; __syncthreads();
  float s3 = bc1[t];
  for (int k = 0; k < HF; ++k) s3 += buf[k] * Wc1[k * HF + t];
  float cv = silu_f(s3) * Wc2[t];
  __syncthreads(); buf[t] = cv; __syncthreads();
  if (t == 0) {
    float sv = bc2[0];
    for (int k = 0; k < HF; ++k) sv += buf[k];
    unsafeAtomicAdd(&agg_c[(size_t)r * 3 + 0], dxyz[0] * sv);
    unsafeAtomicAdd(&agg_c[(size_t)r * 3 + 1], dxyz[1] * sv);
    unsafeAtomicAdd(&agg_c[(size_t)r * 3 + 2], dxyz[2] * sv);
    unsafeAtomicAdd(&cnt[r], 1.0f);
  }
}

__global__ void egcl_node_tail(const float* h, const float* agg_h,
                               const float* Wn1, const float* bn1,
                               const float* Wn2, const float* bn2,
                               float* h_out, int n0, int N)
{
  int node = n0 + blockIdx.x;
  if (node >= N) return;
  __shared__ float nin[2 * NF];
  __shared__ float buf[HF];
  int t = threadIdx.x;
  nin[t]      = h[(size_t)node * NF + t];
  nin[NF + t] = agg_h[(size_t)node * HF + t];
  __syncthreads();
  float s = bn1[t];
  for (int k = 0; k < 2 * NF; ++k) s += nin[k] * Wn1[k * HF + t];
  buf[t] = silu_f(s);
  __syncthreads();
  float s2 = bn2[t];
  for (int k = 0; k < HF; ++k) s2 += buf[k] * Wn2[k * NF + t];
  h_out[(size_t)node * NF + t] = s2;
}

__global__ void coord_out_kernel(const float* coord, const float* agg_c,
                                 const float* cnt, float* out, int N)
{
  int i = blockIdx.x * 256 + threadIdx.x;
  if (i >= N * 3) return;
  float cv = fmaxf(cnt[i / 3], 1.0f);
  out[i] = coord[i] + agg_c[i] / cv;
}

// ---------------------------------------------------------------- launch
extern "C" void kernel_launch(void* const* d_in, const int* in_sizes, int n_in,
                              void* d_out, int out_size, void* d_ws, size_t ws_size,
                              hipStream_t stream)
{
  const float* h     = (const float*)d_in[0];
  const float* coord = (const float*)d_in[1];
  const int*   ei    = (const int*)  d_in[2];
  const float* We1 = (const float*)d_in[3];
  const float* be1 = (const float*)d_in[4];
  const float* We2 = (const float*)d_in[5];
  const float* be2 = (const float*)d_in[6];
  const float* Wn1 = (const float*)d_in[7];
  const float* bn1 = (const float*)d_in[8];
  const float* Wn2 = (const float*)d_in[9];
  const float* bn2 = (const float*)d_in[10];
  const float* Wc1 = (const float*)d_in[11];
  const float* bc1 = (const float*)d_in[12];
  const float* Wc2 = (const float*)d_in[13];
  const float* bc2 = (const float*)d_in[14];

  const int N = in_sizes[0] / NF;
  const int E = in_sizes[2] / 2;

  auto alignup = [](size_t x) { return (x + 255) & ~(size_t)255; };
  char* ws = (char*)d_ws;
  size_t off = 0;
  __bf16* hbf  = (__bf16*)(ws + off); off = alignup(off + (size_t)N * NF * 2);
  __bf16* We1p = (__bf16*)(ws + off); off = alignup(off + 512 * 256 * 2);
  __bf16* We2p = (__bf16*)(ws + off); off = alignup(off + 256 * 256 * 2);
  __bf16* Wc1p = (__bf16*)(ws + off); off = alignup(off + 256 * 256 * 2);
  __bf16* Wn1p = (__bf16*)(ws + off); off = alignup(off + 512 * 256 * 2);
  __bf16* Wn2p = (__bf16*)(ws + off); off = alignup(off + 256 * 256 * 2);
  float* agg_h = (float*)(ws + off);  off = alignup(off + (size_t)N * HF * 4);
  float* agg_c = (float*)(ws + off);  off = alignup(off + (size_t)N * 3 * 4);
  float* cnt   = (float*)(ws + off);  off = alignup(off + (size_t)N * 4);

  float* h_out = (float*)d_out;
  float* c_out = (float*)d_out + (size_t)N * NF;

  // prep: zero accumulators, cvt h to bf16, repack weights to fragment layout
  zero_f32<<<(N * HF + 255) / 256, 256, 0, stream>>>(agg_h, N * HF);
  zero_f32<<<(N * 3 + 255) / 256, 256, 0, stream>>>(agg_c, N * 3);
  zero_f32<<<(N + 255) / 256, 256, 0, stream>>>(cnt, N);
  cvt_bf16<<<(N * NF + 255) / 256, 256, 0, stream>>>(h, hbf, N * NF);
  pack_w<<<(512 * 256 + 255) / 256, 256, 0, stream>>>(We1, We1p, 512);
  pack_w<<<(256 * 256 + 255) / 256, 256, 0, stream>>>(We2, We2p, 256);
  pack_w<<<(256 * 256 + 255) / 256, 256, 0, stream>>>(Wc1, Wc1p, 256);
  pack_w<<<(512 * 256 + 255) / 256, 256, 0, stream>>>(Wn1, Wn1p, 512);
  pack_w<<<(256 * 256 + 255) / 256, 256, 0, stream>>>(Wn2, Wn2p, 256);

  // edge phase
  int etiles = E / 16;
  if (etiles > 0)
    egcl_edge_kernel<<<(etiles + 3) / 4, 128, 0, stream>>>(
        coord, ei, hbf, (const v16bf*)We1p, (const v16bf*)We2p, (const v16bf*)Wc1p,
        We1, be1, be2, bc1, Wc2, bc2, agg_h, agg_c, cnt, E, etiles);
  int etail = E - etiles * 16;
  if (etail > 0)
    egcl_edge_tail<<<etail, 256, 0, stream>>>(h, coord, ei, We1, be1, We2, be2,
                                              Wc1, bc1, Wc2, bc2,
                                              agg_h, agg_c, cnt, etiles * 16, E);

  // node phase
  int ntl = N / 16;
  if (ntl > 0)
    egcl_node_kernel<<<(ntl + 3) / 4, 128, 0, stream>>>(
        hbf, agg_h, (const v16bf*)Wn1p, (const v16bf*)Wn2p, bn1, bn2, h_out, ntl);
  int ntail = N - ntl * 16;
  if (ntail > 0)
    egcl_node_tail<<<ntail, 256, 0, stream>>>(h, agg_h, Wn1, bn1, Wn2, bn2,
                                              h_out, ntl * 16, N);

  coord_out_kernel<<<(N * 3 + 255) / 256, 256, 0, stream>>>(coord, agg_c, cnt, c_out, N);
}
